// HeterGAT_29051158790742
// MI455X (gfx1250) — matrix operand order repaired and checked by reading, not compiled, not run on hardware
//
#include <hip/hip_runtime.h>

#define TT   6
#define NPKG 20000
#define NTGT 50000
#define NE   60000
#define NH   4
#define HCC  256
#define DIN  128
#define EDIM 16

typedef __attribute__((ext_vector_type(16))) __bf16 v16bf;
typedef __attribute__((ext_vector_type(8)))  float  v8f;

// ---------- helpers ----------
__device__ __forceinline__ unsigned f32_ord(float f) {
    unsigned b = __float_as_uint(f);
    return (b & 0x80000000u) ? ~b : (b | 0x80000000u);
}
__device__ __forceinline__ float ord_f32(unsigned u) {
    return (u & 0x80000000u) ? __uint_as_float(u ^ 0x80000000u)
                             : __uint_as_float(~u);
}

// ---------- generic fills ----------
__global__ void fill_f32(float* p, float v, long long n) {
    long long i = (long long)blockIdx.x * blockDim.x + threadIdx.x;
    if (i < n) p[i] = v;
}
__global__ void fill_u32(unsigned* p, unsigned v, long long n) {
    long long i = (long long)blockIdx.x * blockDim.x + threadIdx.x;
    if (i < n) p[i] = v;
}

// ---------- v[k,h] = sum_c W[t,k,64h+c] * a[t,h,c] ----------
__global__ void prep_v(const float* __restrict__ W, const float* __restrict__ A,
                       float* __restrict__ V, int K) {
    int t = blockIdx.x;
    int tid = threadIdx.x;
    if (tid >= K * NH) return;
    int k = tid >> 2, h = tid & 3;
    const float* w = W + ((size_t)t * K + k) * HCC + h * 64;
    const float* a = A + ((size_t)t * NH + h) * 64;
    float s = 0.f;
#pragma unroll 8
    for (int c = 0; c < 64; ++c) s += w[c] * a[c];
    V[((size_t)t * K + k) * NH + h] = s;
}

// ---------- S[t,n,h] = X[(t),n,:K] . V[t,:,h]  (one wave per (t,n)) ----------
__global__ void node_scores(const float* __restrict__ X, long long xTypeStride,
                            const float* __restrict__ V, int K, int N,
                            float* __restrict__ S) {
    long long gtid = (long long)blockIdx.x * blockDim.x + threadIdx.x;
    long long wave = gtid >> 5;
    int lane = (int)(gtid & 31);
    if (wave >= (long long)TT * N) return;
    int t = (int)(wave / N);
    int n = (int)(wave % N);
    const float* x = X + (size_t)t * xTypeStride + (size_t)n * K;
    const float* v = V + (size_t)t * K * NH;
    float p0 = 0.f, p1 = 0.f, p2 = 0.f, p3 = 0.f;
    for (int k = lane; k < K; k += 32) {
        float xv = x[k];
        p0 += xv * v[k * 4 + 0];
        p1 += xv * v[k * 4 + 1];
        p2 += xv * v[k * 4 + 2];
        p3 += xv * v[k * 4 + 3];
    }
#pragma unroll
    for (int o = 16; o; o >>= 1) {
        p0 += __shfl_down(p0, o, 32);
        p1 += __shfl_down(p1, o, 32);
        p2 += __shfl_down(p2, o, 32);
        p3 += __shfl_down(p3, o, 32);
    }
    if (lane == 0) {
        size_t b = ((size_t)t * N + n) * NH;
        S[b + 0] = p0; S[b + 1] = p1; S[b + 2] = p2; S[b + 3] = p3;
    }
}

// ---------- hs[t] = x_pkg @ W[t]  (bf16 WMMA, f32 accum) ----------
// grid: (M/16, 256/64, TT), block: 32 (one wave per 16x64 output tile)
__global__ __launch_bounds__(32)
void gemm_hs_wmma(const float* __restrict__ A, const float* __restrict__ W,
                  float* __restrict__ Out, int M) {
    const int lane = threadIdx.x & 31;
    const int half = lane >> 4;   // 0: lanes 0-15, 1: lanes 16-31
    const int l16  = lane & 15;
    const int row0 = blockIdx.x * 16;
    const int col0 = blockIdx.y * 64;
    const int t    = blockIdx.z;
    const float* Wt = W + (size_t)t * DIN * HCC;
    float* Ot = Out + (size_t)t * M * HCC;
    const float* Arow = A + (size_t)(row0 + l16) * DIN;

    v8f acc0 = {}, acc1 = {}, acc2 = {}, acc3 = {};
#pragma unroll
    for (int kc = 0; kc < DIN; kc += 32) {
        // A fragment: contiguous k order (k = 16*half + i), matching B
        v16bf a;
        const int ka = kc + half * 16;
#pragma unroll
        for (int i = 0; i < 8; ++i) {
            a[i]     = (__bf16)Arow[ka + i];
            a[8 + i] = (__bf16)Arow[ka + 8 + i];
        }
        const int kb = kc + half * 16;
#pragma unroll
        for (int nt = 0; nt < 4; ++nt) {
            const int col = col0 + nt * 16 + l16;
            v16bf b;
#pragma unroll
            for (int i = 0; i < 16; ++i)
                b[i] = (__bf16)Wt[(size_t)(kb + i) * HCC + col];
            v8f& acc = (nt == 0) ? acc0 : (nt == 1) ? acc1 : (nt == 2) ? acc2 : acc3;
            acc = __builtin_amdgcn_wmma_f32_16x16x32_bf16(
                false, a, false, b, (short)0, acc, false, false);
        }
    }
#pragma unroll
    for (int nt = 0; nt < 4; ++nt) {
        v8f acc = (nt == 0) ? acc0 : (nt == 1) ? acc1 : (nt == 2) ? acc2 : acc3;
#pragma unroll
        for (int v = 0; v < 8; ++v) {
            int r = row0 + v + half * 8;
            int c = col0 + nt * 16 + l16;
            Ot[(size_t)r * HCC + c] = acc[v];
        }
    }
}

// ---------- edge logits + leaky relu + segment max ----------
__global__ void edge_logits(const int* __restrict__ EI,
                            const float* __restrict__ Ssrc,
                            const float* __restrict__ Stgt,
                            const float* __restrict__ Sedge,
                            float* __restrict__ Ebuf,
                            unsigned* __restrict__ Menc) {
    long long gid = (long long)blockIdx.x * blockDim.x + threadIdx.x;
    if (gid >= (long long)TT * NE * NH) return;
    int h = (int)(gid & 3);
    long long te = gid >> 2;
    int e = (int)(te % NE);
    int t = (int)(te / NE);
    int src = EI[(size_t)t * 2 * NE + e];
    int dst = EI[(size_t)t * 2 * NE + NE + e];
    float s = Ssrc[((size_t)t * NPKG + src) * NH + h]
            + Stgt[((size_t)t * NTGT + dst) * NH + h]
            + Sedge[gid];
    s = (s > 0.f) ? s : 0.2f * s;   // leaky relu, slope 0.2
    Ebuf[gid] = s;
    atomicMax(&Menc[((size_t)t * NTGT + dst) * NH + h], f32_ord(s));
}

// ---------- exp(e - max) + segment sum ----------
__global__ void edge_exp(const int* __restrict__ EI,
                         float* __restrict__ Ebuf,
                         const unsigned* __restrict__ Menc,
                         float* __restrict__ Den) {
    long long gid = (long long)blockIdx.x * blockDim.x + threadIdx.x;
    if (gid >= (long long)TT * NE * NH) return;
    int h = (int)(gid & 3);
    long long te = gid >> 2;
    int e = (int)(te % NE);
    int t = (int)(te / NE);
    int dst = EI[(size_t)t * 2 * NE + NE + e];
    size_t di = ((size_t)t * NTGT + dst) * NH + h;
    float ex = __expf(Ebuf[gid] - ord_f32(Menc[di]));
    Ebuf[gid] = ex;
    atomicAdd(&Den[di], ex);
}

// ---------- alpha + scatter aggregate: out[dst,:] += hs[src,:] * alpha ----------
__global__ void alpha_agg(const int* __restrict__ EI,
                          const float* __restrict__ Ebuf,
                          const float* __restrict__ Den,
                          const float* __restrict__ HS,
                          float* __restrict__ Outagg,
                          float* __restrict__ AlphaOut) {
    int e = blockIdx.x, t = blockIdx.y, j = threadIdx.x;  // 256 threads
    int src = EI[(size_t)t * 2 * NE + e];
    int dst = EI[(size_t)t * 2 * NE + NE + e];
    int h = j >> 6;
    size_t eb = ((size_t)t * NE + e) * NH;
    size_t db = ((size_t)t * NTGT + dst) * NH;
    float alpha = Ebuf[eb + h] / (Den[db + h] + 1e-16f);
    if (AlphaOut != nullptr && j < NH)
        AlphaOut[eb + j] = Ebuf[eb + j] / (Den[db + j] + 1e-16f);
    atomicAdd(&Outagg[((size_t)t * NTGT + dst) * HCC + j],
              HS[((size_t)t * NPKG + src) * HCC + j] * alpha);
}

// ---------- +bias, LayerNorm+ReLU (mask: t==5 bypass), optional h-store,
//            optional fused s_tgt2 = h . vtgt2 ----------
__global__ void ln_relu(const float* __restrict__ In, const float* __restrict__ Bias,
                        const float* __restrict__ G, const float* __restrict__ Be,
                        const float* __restrict__ Vt,  // [T,256,4] or null
                        float* __restrict__ Hout,      // null -> don't store h
                        float* __restrict__ Stgt) {    // null -> no score
    int n = blockIdx.x, t = blockIdx.y, j = threadIdx.x;
    __shared__ float red[HCC];
    size_t base = ((size_t)t * NTGT + n) * HCC;
    float x = In[base + j] + Bias[t * HCC + j];
    red[j] = x; __syncthreads();
    for (int s = 128; s; s >>= 1) { if (j < s) red[j] += red[j + s]; __syncthreads(); }
    float mu = red[0] * (1.f / HCC); __syncthreads();
    float d = x - mu;
    red[j] = d * d; __syncthreads();
    for (int s = 128; s; s >>= 1) { if (j < s) red[j] += red[j + s]; __syncthreads(); }
    float var = red[0] * (1.f / HCC); __syncthreads();
    float y = d * rsqrtf(var + 1e-5f) * G[t * HCC + j] + Be[t * HCC + j];
    y = fmaxf(y, 0.f);
    float hval = (t == 5) ? x : y;   // LN_MASK: 'Hostnames' bypass
    if (Hout) Hout[base + j] = hval;
    if (Stgt) {
#pragma unroll
        for (int h = 0; h < NH; ++h) {
            red[j] = hval * Vt[((size_t)t * HCC + j) * NH + h]; __syncthreads();
            for (int s = 128; s; s >>= 1) { if (j < s) red[j] += red[j + s]; __syncthreads(); }
            if (j == 0) Stgt[((size_t)t * NTGT + n) * NH + h] = red[0];
            __syncthreads();
        }
    }
}

// ---------- pooled column sums ----------
__global__ void colsum_h2(const float* __restrict__ H2, float* __restrict__ EmbSum) {
    int t = blockIdx.x, chunk = blockIdx.y, j = threadIdx.x;  // 256 threads
    float s = 0.f;
    for (int n = chunk; n < NTGT; n += gridDim.y)
        s += H2[((size_t)t * NTGT + n) * HCC + j];
    atomicAdd(&EmbSum[t * HCC + j], s);
}
__global__ void colsum_pkg(const float* __restrict__ X, float* __restrict__ EmbSum) {
    int chunk = blockIdx.x, j = threadIdx.x;  // 128 threads
    float s = 0.f;
    for (int n = chunk; n < NPKG; n += gridDim.x)
        s += X[(size_t)n * DIN + j];
    atomicAdd(&EmbSum[TT * HCC + j], s);
}

// ---------- final logit ----------
__global__ void logits_k(const float* __restrict__ EmbSum, const float* __restrict__ Wc,
                         const float* __restrict__ bc, float* __restrict__ Out0) {
    __shared__ float red[256];
    int j = threadIdx.x;
    float s = 0.f;
    for (int i = j; i < TT * HCC + DIN; i += 256) {
        float m = EmbSum[i] * ((i < TT * HCC) ? (1.f / NTGT) : (1.f / NPKG));
        s += m * Wc[i];
    }
    red[j] = s; __syncthreads();
    for (int t = 128; t; t >>= 1) { if (j < t) red[j] += red[j + t]; __syncthreads(); }
    if (j == 0) Out0[0] = red[0] + bc[0];
}

// =====================================================================
extern "C" void kernel_launch(void* const* d_in, const int* in_sizes, int n_in,
                              void* d_out, int out_size, void* d_ws, size_t ws_size,
                              hipStream_t stream) {
    (void)in_sizes; (void)n_in; (void)out_size; (void)ws_size;
    const float* x_pkg = (const float*)d_in[0];
    const float* x_tgt = (const float*)d_in[1];
    const int*   ei    = (const int*)d_in[2];
    const float* eattr = (const float*)d_in[3];
    const float* Wsrc1 = (const float*)d_in[4];
    const float* Wtgt1 = (const float*)d_in[5];
    const float* Wedge1= (const float*)d_in[6];
    const float* as1   = (const float*)d_in[7];
    const float* at1   = (const float*)d_in[8];
    const float* ae1   = (const float*)d_in[9];
    const float* b1    = (const float*)d_in[10];
    const float* g1    = (const float*)d_in[11];
    const float* be1   = (const float*)d_in[12];
    const float* Wsrc2 = (const float*)d_in[13];
    const float* Wtgt2 = (const float*)d_in[14];
    const float* Wedge2= (const float*)d_in[15];
    const float* as2   = (const float*)d_in[16];
    const float* at2   = (const float*)d_in[17];
    const float* ae2   = (const float*)d_in[18];
    const float* b2    = (const float*)d_in[19];
    const float* Wc    = (const float*)d_in[20];
    const float* bc    = (const float*)d_in[21];
    float* out = (float*)d_out;           // out[0]=logits, out[1..]=alpha2[T,E,H]

    // ---- workspace carve-up (floats) ----
    float* ws = (float*)d_ws;
    size_t off = 0;
    float* hs     = ws + off; off += (size_t)TT * NPKG * HCC;   // hs1 then hs2
    float* outb   = ws + off; off += (size_t)TT * NTGT * HCC;   // out1 -> (zero) -> out2/h2
    float* ssrc   = ws + off; off += (size_t)TT * NPKG * NH;
    float* stgt   = ws + off; off += (size_t)TT * NTGT * NH;
    float* sedge  = ws + off; off += (size_t)TT * NE * NH;
    float* ebuf   = ws + off; off += (size_t)TT * NE * NH;
    unsigned* menc = (unsigned*)(ws + off); off += (size_t)TT * NTGT * NH;
    float* denom  = ws + off; off += (size_t)TT * NTGT * NH;
    float* vsrc1  = ws + off; off += (size_t)TT * DIN * NH;
    float* vtgt1  = ws + off; off += (size_t)TT * DIN * NH;
    float* vedge1 = ws + off; off += (size_t)TT * EDIM * NH;
    float* vsrc2  = ws + off; off += (size_t)TT * DIN * NH;
    float* vtgt2  = ws + off; off += (size_t)TT * HCC * NH;
    float* vedge2 = ws + off; off += (size_t)TT * EDIM * NH;
    float* embsum = ws + off; off += (size_t)(TT * HCC + DIN);

    const long long nNodeHT = (long long)TT * NTGT * NH;
    const long long nOut    = (long long)TT * NTGT * HCC;
    const long long nEH     = (long long)TT * NE * NH;
    auto blk = [](long long n, int b) { return (unsigned)((n + b - 1) / b); };

    // ---- attention projection vectors ----
    prep_v<<<TT, DIN * NH, 0, stream>>>(Wsrc1, as1, vsrc1, DIN);
    prep_v<<<TT, DIN * NH, 0, stream>>>(Wtgt1, at1, vtgt1, DIN);
    prep_v<<<TT, EDIM * NH, 0, stream>>>(Wedge1, ae1, vedge1, EDIM);
    prep_v<<<TT, DIN * NH, 0, stream>>>(Wsrc2, as2, vsrc2, DIN);
    prep_v<<<TT, HCC * NH, 0, stream>>>(Wtgt2, at2, vtgt2, HCC);
    prep_v<<<TT, EDIM * NH, 0, stream>>>(Wedge2, ae2, vedge2, EDIM);

    // ======================= layer 1 =======================
    node_scores<<<blk((long long)TT * NPKG * 32, 256), 256, 0, stream>>>(
        x_pkg, 0LL, vsrc1, DIN, NPKG, ssrc);
    node_scores<<<blk((long long)TT * NTGT * 32, 256), 256, 0, stream>>>(
        x_tgt, (long long)NTGT * DIN, vtgt1, DIN, NTGT, stgt);
    node_scores<<<blk((long long)TT * NE * 32, 256), 256, 0, stream>>>(
        eattr, (long long)NE * EDIM, vedge1, EDIM, NE, sedge);

    gemm_hs_wmma<<<dim3(NPKG / 16, HCC / 64, TT), 32, 0, stream>>>(x_pkg, Wsrc1, hs, NPKG);

    fill_u32<<<blk(nNodeHT, 256), 256, 0, stream>>>(menc, 0u, nNodeHT);
    fill_f32<<<blk(nNodeHT, 256), 256, 0, stream>>>(denom, 0.f, nNodeHT);
    fill_f32<<<blk(nOut, 256), 256, 0, stream>>>(outb, 0.f, nOut);

    edge_logits<<<blk(nEH, 256), 256, 0, stream>>>(ei, ssrc, stgt, sedge, ebuf, menc);
    edge_exp<<<blk(nEH, 256), 256, 0, stream>>>(ei, ebuf, menc, denom);
    alpha_agg<<<dim3(NE, TT), HCC, 0, stream>>>(ei, ebuf, denom, hs, outb, nullptr);

    // LN+ReLU of out1 fused with s_tgt2 = h1 . vtgt2 (h1 never materialized)
    ln_relu<<<dim3(NTGT, TT), HCC, 0, stream>>>(outb, b1, g1, be1, vtgt2, nullptr, stgt);

    // ======================= layer 2 =======================
    node_scores<<<blk((long long)TT * NPKG * 32, 256), 256, 0, stream>>>(
        x_pkg, 0LL, vsrc2, DIN, NPKG, ssrc);
    node_scores<<<blk((long long)TT * NE * 32, 256), 256, 0, stream>>>(
        eattr, (long long)NE * EDIM, vedge2, EDIM, NE, sedge);

    gemm_hs_wmma<<<dim3(NPKG / 16, HCC / 64, TT), 32, 0, stream>>>(x_pkg, Wsrc2, hs, NPKG);

    fill_u32<<<blk(nNodeHT, 256), 256, 0, stream>>>(menc, 0u, nNodeHT);
    fill_f32<<<blk(nNodeHT, 256), 256, 0, stream>>>(denom, 0.f, nNodeHT);
    fill_f32<<<blk(nOut, 256), 256, 0, stream>>>(outb, 0.f, nOut);

    edge_logits<<<blk(nEH, 256), 256, 0, stream>>>(ei, ssrc, stgt, sedge, ebuf, menc);
    edge_exp<<<blk(nEH, 256), 256, 0, stream>>>(ei, ebuf, menc, denom);
    alpha_agg<<<dim3(NE, TT), HCC, 0, stream>>>(ei, ebuf, denom, hs, outb, out + 1);

    // LN (bug-preserving g1/be1 reuse), h2 written in place over outb
    ln_relu<<<dim3(NTGT, TT), HCC, 0, stream>>>(outb, b2, g1, be1, nullptr, outb, nullptr);

    // ======================= pooling + classifier =======================
    fill_f32<<<blk(TT * HCC + DIN, 256), 256, 0, stream>>>(embsum, 0.f, TT * HCC + DIN);
    colsum_h2<<<dim3(TT, 64), HCC, 0, stream>>>(outb, embsum);
    colsum_pkg<<<64, DIN, 0, stream>>>(x_pkg, embsum);
    logits_k<<<1, 256, 0, stream>>>(embsum, Wc, bc, out);
}
